// GCN_29265907155154
// MI455X (gfx1250) — matrix operand order repaired and checked by reading, not compiled, not run on hardware
//
#include <hip/hip_runtime.h>
#include <hip/hip_bf16.h>

// ---------------------------------------------------------------------------
// GCN (5x GCNConv + BN + ReLU) for MI455X / gfx1250, wave32.
//
//  * GEMMs on the WMMA pipe: f32 activations split into bf16 hi + residual,
//    D = Ahi*Bhi + Ahi*Blo + Alo*Bhi  (3x v_wmma_f32_16x16x32_bf16 per tile)
//    => ~f32 precision at bf16-WMMA rate. Weights pre-packed once per launch
//    into WMMA B-fragment lane order (bf16 hi/lo). GEMM shapes are template
//    parameters so the whole reduction is straight-line loads + WMMA.
//  * Edge aggregation (the bottleneck, ~6 GB of traffic) uses one wave32 per
//    edge: precomputed per-edge norm, coalesced 128 B gathers of h[src],
//    global_atomic_add_f32 scatter into agg[dst].
//  * Self-loop + bias + BN(eval) + ReLU fused into one elementwise kernel,
//    written in place so the agg buffer becomes the next layer's input.
// ---------------------------------------------------------------------------

typedef __attribute__((ext_vector_type(16))) __bf16 v16bf;
typedef __attribute__((ext_vector_type(8)))  float  v8f;

#define GCN_N 100000
#define GCN_E 1600000
#define BN_EPS 1e-5f

// ---------------------------------------------------------------------------
__global__ __launch_bounds__(256) void k_zero(float* __restrict__ p, int n) {
    int i = blockIdx.x * blockDim.x + threadIdx.x;
    if (i < n) p[i] = 0.0f;
}

// Degree at destination nodes (edge weight 1).
__global__ __launch_bounds__(256) void k_degree(const int* __restrict__ dst,
                                                float* __restrict__ deg, int E) {
    int e = blockIdx.x * blockDim.x + threadIdx.x;
    if (e < E) atomicAdd(&deg[dst[e]], 1.0f);
}

// deg -> rsqrt(deg + 1)  (self-loop included), in place.
__global__ __launch_bounds__(256) void k_dinv(float* __restrict__ deg, int n) {
    int i = blockIdx.x * blockDim.x + threadIdx.x;
    if (i < n) deg[i] = rsqrtf(deg[i] + 1.0f);
}

// Per-edge symmetric norm (hoisted out of all 5 scatter passes).
__global__ __launch_bounds__(256) void k_edge_norm(
        const int* __restrict__ src, const int* __restrict__ dst,
        const float* __restrict__ dinv, float* __restrict__ norm, int E) {
    int e = blockIdx.x * blockDim.x + threadIdx.x;
    if (e < E) norm[e] = dinv[src[e]] * dinv[dst[e]];
}

// ---------------------------------------------------------------------------
// Pack W (KxM f32 row-major) into WMMA B-fragment lane order, bf16 hi + lo.
// Fragment element index: ((kc*nct + ct)*32 + lane)*16 + e
//   value = W[(kc*32 + (e>>3)*16 + (lane>>4)*8 + (e&7))*M + ct*16 + (lane&15)]   (0 beyond M)
__global__ __launch_bounds__(256) void k_pack_w(
        const float* __restrict__ W, __bf16* __restrict__ hi,
        __bf16* __restrict__ lo, int K, int M, int nct) {
    int idx = blockIdx.x * blockDim.x + threadIdx.x;
    int total = (K / 32) * nct * 32 * 16;
    if (idx >= total) return;
    int e    = idx & 15;
    int lane = (idx >> 4) & 31;
    int rest = idx >> 9;                 // kc*nct + ct
    int ct   = rest % nct;
    int kc   = rest / nct;
    int k = kc * 32 + (e >> 3) * 16 + (lane >> 4) * 8 + (e & 7);
    int n = ct * 16 + (lane & 15);
    float w = (n < M) ? W[(size_t)k * M + n] : 0.0f;
    __bf16 h = (__bf16)w;
    hi[idx] = h;
    lo[idx] = (__bf16)(w - (float)h);
}

// ---------------------------------------------------------------------------
// WMMA GEMM:  H[NrowsxM] = A[NrowsxK] @ W[KxM]   (f32 activations in/out,
// weights pre-packed bf16 fragments). Compile-time K, M => fully unrolled,
// branch-free reduction. One wave per 16-row tile.
template <int K, int M>
__global__ __launch_bounds__(256) void k_wmma_gemm(
        const float* __restrict__ A, const __bf16* __restrict__ Whi,
        const __bf16* __restrict__ Wlo, float* __restrict__ H, int Nrows) {
    constexpr int NCT = (M + 15) / 16;
    constexpr int NKC = K / 32;

    const int wave = threadIdx.x >> 5;
    const int lane = threadIdx.x & 31;
    const int rowTile = blockIdx.x * 8 + wave;
    if (rowTile * 16 >= Nrows) return;      // whole-wave exit: EXEC stays all-1s

    const int m    = lane & 15;             // A: row within tile / C: column
    const int half = lane >> 4;

    const float* arow = A + (size_t)(rowTile * 16 + m) * K;
    const v16bf* Bhi  = (const v16bf*)Whi + lane;
    const v16bf* Blo  = (const v16bf*)Wlo + lane;

    v8f acc[NCT];
#pragma unroll
    for (int ct = 0; ct < NCT; ++ct) acc[ct] = (v8f){0,0,0,0,0,0,0,0};

#pragma unroll
    for (int kc = 0; kc < NKC; ++kc) {
        const int k0 = kc * 32;
        // ---- A fragment (16x32 bf16, ISA 7.12.2): lane(16*half+m)
        //      elems 0..7  -> K = k0 +      half*8 + e
        //      elems 8..15 -> K = k0 + 16 + half*8 + (e-8)
        v16bf ahi, alo;
#pragma unroll
        for (int j = 0; j < 8; ++j) {
            float f0 = arow[k0 + half * 8 + j];
            float f1 = arow[k0 + 16 + half * 8 + j];
            __bf16 h0 = (__bf16)f0;
            __bf16 h1 = (__bf16)f1;
            ahi[j]     = h0;  alo[j]     = (__bf16)(f0 - (float)h0);
            ahi[8 + j] = h1;  alo[8 + j] = (__bf16)(f1 - (float)h1);
        }
#pragma unroll
        for (int ct = 0; ct < NCT; ++ct) {
            v16bf bhi = Bhi[(kc * NCT + ct) * 32];
            v16bf blo = Blo[(kc * NCT + ct) * 32];
            acc[ct] = __builtin_amdgcn_wmma_f32_16x16x32_bf16(
                false, ahi, false, bhi, (short)0, acc[ct], false, false);
            acc[ct] = __builtin_amdgcn_wmma_f32_16x16x32_bf16(
                false, ahi, false, blo, (short)0, acc[ct], false, false);
            acc[ct] = __builtin_amdgcn_wmma_f32_16x16x32_bf16(
                false, alo, false, bhi, (short)0, acc[ct], false, false);
        }
    }

    // ---- store C: lane(16*half+n) holds column n, VGPR i -> row half*8 + i
    const int rbase = rowTile * 16 + half * 8;
#pragma unroll
    for (int ct = 0; ct < NCT; ++ct) {
        const int n0 = ct * 16;
        if ((M % 16 == 0) || (n0 + m < M)) {   // compile-time unless M==2
#pragma unroll
            for (int i = 0; i < 8; ++i)
                H[(size_t)(rbase + i) * M + n0 + m] = acc[ct][i];
        }
    }
}

// ---------------------------------------------------------------------------
// Edge scatter: one wave32 per edge.
//   agg[dst] += norm[e] * h[src]
template <int F>
__global__ __launch_bounds__(256) void k_scatter(
        const float* __restrict__ h, const int* __restrict__ src,
        const int* __restrict__ dst, const float* __restrict__ norm,
        float* __restrict__ agg, int E) {
    const int e    = (blockIdx.x * blockDim.x + threadIdx.x) >> 5;
    const int lane = threadIdx.x & 31;
    if (e >= E) return;
    const int s = src[e];
    const int d = dst[e];
    const float w = norm[e];
    const float* hs = h + (size_t)s * F;
    float*       ad = agg + (size_t)d * F;
#pragma unroll
    for (int c = lane; c < F; c += 32)
        atomicAdd(&ad[c], w * hs[c]);
}

// ---------------------------------------------------------------------------
// Fused: self-loop + bias + BatchNorm(eval) + ReLU, in place in agg.
template <int F>
__global__ __launch_bounds__(256) void k_finalize_bn_relu(
        float* __restrict__ agg, const float* __restrict__ h,
        const float* __restrict__ dinv, const float* __restrict__ bias,
        const float* __restrict__ g, const float* __restrict__ be,
        const float* __restrict__ rm, const float* __restrict__ rv, int Nn) {
    int tid = blockIdx.x * blockDim.x + threadIdx.x;
    if (tid >= Nn * F) return;
    int n = tid / F, c = tid % F;
    float di = dinv[n];
    float v = agg[tid] + di * di * h[tid] + bias[c];
    v = (v - rm[c]) * rsqrtf(rv[c] + BN_EPS) * g[c] + be[c];
    agg[tid] = fmaxf(v, 0.0f);
}

// Last layer: self-loop + bias only, write to output.
template <int F>
__global__ __launch_bounds__(256) void k_finalize_last(
        const float* __restrict__ agg, const float* __restrict__ h,
        const float* __restrict__ dinv, const float* __restrict__ bias,
        float* __restrict__ out, int Nn) {
    int tid = blockIdx.x * blockDim.x + threadIdx.x;
    if (tid >= Nn * F) return;
    int n = tid / F, c = tid % F;
    float di = dinv[n];
    out[tid] = agg[tid] + di * di * h[tid] + bias[c];
}

// ---------------------------------------------------------------------------
extern "C" void kernel_launch(void* const* d_in, const int* in_sizes, int n_in,
                              void* d_out, int out_size, void* d_ws, size_t ws_size,
                              hipStream_t stream) {
    (void)in_sizes; (void)n_in; (void)out_size; (void)ws_size;
    const int N = GCN_N, E = GCN_E;

    const float* x    = (const float*)d_in[0];
    const int*   ei   = (const int*)d_in[1];
    const int*   srcp = ei;
    const int*   dstp = ei + E;
    const float* Wl[5] = {(const float*)d_in[2], (const float*)d_in[4],
                          (const float*)d_in[6], (const float*)d_in[8],
                          (const float*)d_in[10]};
    const float* bl[5] = {(const float*)d_in[3], (const float*)d_in[5],
                          (const float*)d_in[7], (const float*)d_in[9],
                          (const float*)d_in[11]};
    const float* gl[4]  = {(const float*)d_in[12], (const float*)d_in[16],
                           (const float*)d_in[20], (const float*)d_in[24]};
    const float* bel[4] = {(const float*)d_in[13], (const float*)d_in[17],
                           (const float*)d_in[21], (const float*)d_in[25]};
    const float* rml[4] = {(const float*)d_in[14], (const float*)d_in[18],
                           (const float*)d_in[22], (const float*)d_in[26]};
    const float* rvl[4] = {(const float*)d_in[15], (const float*)d_in[19],
                           (const float*)d_in[23], (const float*)d_in[27]};

    const int Kdim[5] = {128, 128, 128, 128, 64};
    const int Mdim[5] = {128, 128, 128, 64, 2};

    // workspace: dinv | norm | h | bufA | bufB | packed weights (bf16 hi/lo)
    float* dinv = (float*)d_ws;
    float* norm = dinv + N;
    float* hbuf = norm + E;
    float* bufA = hbuf + (size_t)N * 128;
    float* bufB = bufA + (size_t)N * 128;
    __bf16* wpk = (__bf16*)(bufB + (size_t)N * 128);
    __bf16* whi[5]; __bf16* wlo[5]; int wfrag[5];
    {
        __bf16* p = wpk;
        for (int l = 0; l < 5; ++l) {
            int nct = (Mdim[l] + 15) / 16;
            wfrag[l] = (Kdim[l] / 32) * nct * 32 * 16;   // fragment elements
            whi[l] = p; p += wfrag[l];
            wlo[l] = p; p += wfrag[l];
        }
    }

    const int T = 256;
    const int gemmBlocks = (N / 16 + 7) / 8;             // 6250 row tiles / 8 waves
    const int scatBlocks = (E * 32 + T - 1) / T;         // one wave per edge
    #define CDIV(a) (((a) + T - 1) / T)

    // --- one-time prep: degrees -> dinv, edge norms, packed weights
    k_zero<<<CDIV(N), T, 0, stream>>>(dinv, N);
    k_degree<<<CDIV(E), T, 0, stream>>>(dstp, dinv, E);
    k_dinv<<<CDIV(N), T, 0, stream>>>(dinv, N);
    k_edge_norm<<<CDIV(E), T, 0, stream>>>(srcp, dstp, dinv, norm, E);
    for (int l = 0; l < 5; ++l) {
        int nct = (Mdim[l] + 15) / 16;
        k_pack_w<<<CDIV(wfrag[l]), T, 0, stream>>>(
            Wl[l], whi[l], wlo[l], Kdim[l], Mdim[l], nct);
    }

    // --- layer 1: x(128) -> bufA(128)
    k_wmma_gemm<128, 128><<<gemmBlocks, T, 0, stream>>>(x, whi[0], wlo[0], hbuf, N);
    k_zero<<<CDIV(N * 128), T, 0, stream>>>(bufA, N * 128);
    k_scatter<128><<<scatBlocks, T, 0, stream>>>(hbuf, srcp, dstp, norm, bufA, E);
    k_finalize_bn_relu<128><<<CDIV(N * 128), T, 0, stream>>>(
        bufA, hbuf, dinv, bl[0], gl[0], bel[0], rml[0], rvl[0], N);

    // --- layer 2: bufA(128) -> bufB(128)
    k_wmma_gemm<128, 128><<<gemmBlocks, T, 0, stream>>>(bufA, whi[1], wlo[1], hbuf, N);
    k_zero<<<CDIV(N * 128), T, 0, stream>>>(bufB, N * 128);
    k_scatter<128><<<scatBlocks, T, 0, stream>>>(hbuf, srcp, dstp, norm, bufB, E);
    k_finalize_bn_relu<128><<<CDIV(N * 128), T, 0, stream>>>(
        bufB, hbuf, dinv, bl[1], gl[1], bel[1], rml[1], rvl[1], N);

    // --- layer 3: bufB(128) -> bufA(128)
    k_wmma_gemm<128, 128><<<gemmBlocks, T, 0, stream>>>(bufB, whi[2], wlo[2], hbuf, N);
    k_zero<<<CDIV(N * 128), T, 0, stream>>>(bufA, N * 128);
    k_scatter<128><<<scatBlocks, T, 0, stream>>>(hbuf, srcp, dstp, norm, bufA, E);
    k_finalize_bn_relu<128><<<CDIV(N * 128), T, 0, stream>>>(
        bufA, hbuf, dinv, bl[2], gl[2], bel[2], rml[2], rvl[2], N);

    // --- layer 4: bufA(128) -> bufB(64)
    k_wmma_gemm<128, 64><<<gemmBlocks, T, 0, stream>>>(bufA, whi[3], wlo[3], hbuf, N);
    k_zero<<<CDIV(N * 64), T, 0, stream>>>(bufB, N * 64);
    k_scatter<64><<<scatBlocks, T, 0, stream>>>(hbuf, srcp, dstp, norm, bufB, E);
    k_finalize_bn_relu<64><<<CDIV(N * 64), T, 0, stream>>>(
        bufB, hbuf, dinv, bl[3], gl[3], bel[3], rml[3], rvl[3], N);

    // --- layer 5: bufB(64) -> d_out(2)
    k_wmma_gemm<64, 2><<<gemmBlocks, T, 0, stream>>>(bufB, whi[4], wlo[4], hbuf, N);
    k_zero<<<CDIV(N * 2), T, 0, stream>>>(bufA, N * 2);
    k_scatter<2><<<scatBlocks, T, 0, stream>>>(hbuf, srcp, dstp, norm, bufA, E);
    k_finalize_last<2><<<CDIV(N * 2), T, 0, stream>>>(
        bufA, hbuf, dinv, bl[4], (float*)d_out, N);

    #undef CDIV
}
